// VQGAN4REC_81020263071804
// MI455X (gfx1250) — matrix operand order repaired and checked by reading, not compiled, not run on hardware
//
#include <hip/hip_runtime.h>

// ---------------------------------------------------------------------------
// VQ-VAE-style recommender forward for MI455X (gfx1250, wave32, WMMA).
// Large GEMMs: v_wmma_f32_16x16x32_bf16 with async global->LDS double buffering.
// ---------------------------------------------------------------------------

typedef __attribute__((ext_vector_type(16))) __bf16 v16bf;
typedef __attribute__((ext_vector_type(8)))  float  v8f;

#define BM 128
#define BN 128
#define BK 32
#define LDSPAD 8
#define TPB 256

__device__ __forceinline__ __bf16 f2bf(float f) {
    unsigned u = __float_as_uint(f);
    u += 0x7FFFu + ((u >> 16) & 1u);          // round-to-nearest-even
    unsigned short s = (unsigned short)(u >> 16);
    union { unsigned short s; __bf16 b; } cv; cv.s = s;
    return cv.b;
}

// CDNA5 async copy: per-lane 16B global -> LDS, tracked with ASYNCcnt.
// LDS operand is the 32-bit LDS byte address (low 32 bits of flat pointer).
__device__ __forceinline__ void async_ld_b128(unsigned lds, const void* g) {
    asm volatile("global_load_async_to_lds_b128 %0, %1, off"
                 :: "v"(lds), "v"((unsigned long long)(uintptr_t)g)
                 : "memory");
}
__device__ __forceinline__ void wait_async0() {
    asm volatile("s_wait_asynccnt 0x0" ::: "memory");
}

// ---------------------------------------------------------------------------
// Tiled GEMM: C[M,N] = A[M,K] (bf16, row-major) * Bt[N,K] (bf16) + bias
// VQ=true: instead of storing C, compute score = cnorm[col] - 2*acc and
// reduce argmin per row into minbuf (packed u64: float-ordered key | col).
// ---------------------------------------------------------------------------
template <bool VQ>
__global__ __launch_bounds__(TPB)
void k_gemm(const __bf16* __restrict__ A, const __bf16* __restrict__ Bt,
            const float* __restrict__ bias, float* __restrict__ C,
            unsigned long long* __restrict__ minbuf,
            const float* __restrict__ cnorm,
            int M, int N, int K) {
    __shared__ __bf16 As[2][BM][BK + LDSPAD];
    __shared__ __bf16 Bs[2][BN][BK + LDSPAD];

    const int tid  = threadIdx.x;
    const int lane = tid & 31;
    const int wid  = tid >> 5;      // 0..7
    const int wrow = wid >> 2;      // 0..1 : 64 rows each
    const int wcol = wid & 3;       // 0..3 : 32 cols each
    const int half = lane >> 4;     // 0/1
    const int l16  = lane & 15;

    const int blockRow = blockIdx.y * BM;
    const int blockCol = blockIdx.x * BN;

    // per-thread staging slots: 2 x 16B for A tile, 2 x 16B for B tile
    const int i0 = tid;             // 0..255
    const int i1 = tid + TPB;       // 256..511
    const int m0 = i0 >> 2, k0 = (i0 & 3) * 8;
    const int m1 = i1 >> 2, k1 = (i1 & 3) * 8;
    const __bf16* gA0 = A  + (size_t)(blockRow + m0) * K + k0;
    const __bf16* gA1 = A  + (size_t)(blockRow + m1) * K + k1;
    const __bf16* gB0 = Bt + (size_t)(blockCol + m0) * K + k0;
    const __bf16* gB1 = Bt + (size_t)(blockCol + m1) * K + k1;
    unsigned lA0[2], lA1[2], lB0[2], lB1[2];
#pragma unroll
    for (int b = 0; b < 2; ++b) {
        lA0[b] = (unsigned)(uintptr_t)&As[b][m0][k0];
        lA1[b] = (unsigned)(uintptr_t)&As[b][m1][k1];
        lB0[b] = (unsigned)(uintptr_t)&Bs[b][m0][k0];
        lB1[b] = (unsigned)(uintptr_t)&Bs[b][m1][k1];
    }

    v8f acc[4][2];
#pragma unroll
    for (int mi = 0; mi < 4; ++mi)
#pragma unroll
        for (int ni = 0; ni < 2; ++ni)
#pragma unroll
            for (int e = 0; e < 8; ++e) acc[mi][ni][e] = 0.f;

    union Frag { v16bf v; unsigned u[8]; };

    // prologue: stage tile 0 into buffer 0
    async_ld_b128(lA0[0], gA0);
    async_ld_b128(lA1[0], gA1);
    async_ld_b128(lB0[0], gB0);
    async_ld_b128(lB1[0], gB1);
    wait_async0();
    __syncthreads();

    const int nt = K / BK;
    for (int t = 0; t < nt; ++t) {
        const int cur = t & 1;
        // overlap: DMA next K-tile into the other buffer while computing
        if (t + 1 < nt) {
            const int nxt = cur ^ 1;
            const int ko  = (t + 1) * BK;
            async_ld_b128(lA0[nxt], gA0 + ko);
            async_ld_b128(lA1[nxt], gA1 + ko);
            async_ld_b128(lB0[nxt], gB0 + ko);
            async_ld_b128(lB1[nxt], gB1 + ko);
        }

        // ---- build fragments per CDNA5 WMMA VGPR layouts ----
        Frag a[4], b[2];
#pragma unroll
        for (int mi = 0; mi < 4; ++mi) {
            int r = wrow * 64 + mi * 16 + l16;   // A: lane = M row
#pragma unroll
            for (int v = 0; v < 8; ++v) {
                int k = (v >> 2) * 16 + half * 8 + (v & 3) * 2;  // A K-pair layout
                a[mi].u[v] = *(const unsigned*)&As[cur][r][k];
            }
        }
#pragma unroll
        for (int ni = 0; ni < 2; ++ni) {
            int c = wcol * 32 + ni * 16 + l16;   // B: lane = N col
#pragma unroll
            for (int v = 0; v < 8; ++v) {
                int k = (v >> 2) * 16 + half * 8 + (v & 3) * 2;  // B K-pair layout (same as A)
                b[ni].u[v] = *(const unsigned*)&Bs[cur][c][k];
            }
        }

#pragma unroll
        for (int mi = 0; mi < 4; ++mi)
#pragma unroll
            for (int ni = 0; ni < 2; ++ni)
                acc[mi][ni] = __builtin_amdgcn_wmma_f32_16x16x32_bf16(
                    false, a[mi].v, false, b[ni].v, (short)0, acc[mi][ni],
                    false, false);

        // next tile's DMA must land and all waves finish reading cur buffer
        wait_async0();
        __syncthreads();
    }

    if (!VQ) {
        // ---- C layout: lanes 0-15 -> M=v, lanes 16-31 -> M=v+8; N=l16 ----
#pragma unroll
        for (int mi = 0; mi < 4; ++mi)
#pragma unroll
            for (int ni = 0; ni < 2; ++ni) {
                int col  = blockCol + wcol * 32 + ni * 16 + l16;
                float bv = bias ? bias[col] : 0.f;
#pragma unroll
                for (int v = 0; v < 8; ++v) {
                    int row = blockRow + wrow * 64 + mi * 16 + half * 8 + v;
                    C[(size_t)row * N + col] = acc[mi][ni][v] + bv;
                }
            }
    } else {
        // ---- fused VQ argmin epilogue ----
#pragma unroll
        for (int mi = 0; mi < 4; ++mi)
#pragma unroll
            for (int v = 0; v < 8; ++v) {
                unsigned long long best = ~0ull;
#pragma unroll
                for (int ni = 0; ni < 2; ++ni) {
                    int col = blockCol + wcol * 32 + ni * 16 + l16;
                    float s = cnorm[col] - 2.f * acc[mi][ni][v];
                    unsigned ku = __float_as_uint(s);
                    ku = (ku & 0x80000000u) ? ~ku : (ku | 0x80000000u);
                    unsigned long long p =
                        ((unsigned long long)ku << 32) | (unsigned)col;
                    if (p < best) best = p;
                }
                // lanes {0..15} (and {16..31}) share a row: xor-shuffle min
#pragma unroll
                for (int msk = 1; msk < 16; msk <<= 1) {
                    unsigned lo  = (unsigned)best, hi = (unsigned)(best >> 32);
                    unsigned olo = __shfl_xor(lo, msk, 32);
                    unsigned ohi = __shfl_xor(hi, msk, 32);
                    unsigned long long o = ((unsigned long long)ohi << 32) | olo;
                    if (o < best) best = o;
                }
                if (l16 == 0) {
                    int row = blockRow + wrow * 64 + mi * 16 + half * 8 + v;
                    atomicMin(&minbuf[row], best);
                }
            }
    }
}

// ---------------------------------------------------------------------------
// LayerNorm over last dim + residual, output bf16 (next GEMM's A operand).
// ---------------------------------------------------------------------------
__global__ __launch_bounds__(256)
void k_ln_res(const float* __restrict__ H, const float* __restrict__ g,
              const float* __restrict__ be, __bf16* __restrict__ out, int N) {
    __shared__ float red[256];
    const int row = blockIdx.x;
    const float* x = H + (size_t)row * N;
    float s = 0.f;
    for (int j = threadIdx.x; j < N; j += 256) s += x[j];
    red[threadIdx.x] = s; __syncthreads();
    for (int o = 128; o > 0; o >>= 1) {
        if (threadIdx.x < o) red[threadIdx.x] += red[threadIdx.x + o];
        __syncthreads();
    }
    float mu = red[0] / N;
    __syncthreads();
    float vv = 0.f;
    for (int j = threadIdx.x; j < N; j += 256) { float d = x[j] - mu; vv += d * d; }
    red[threadIdx.x] = vv; __syncthreads();
    for (int o = 128; o > 0; o >>= 1) {
        if (threadIdx.x < o) red[threadIdx.x] += red[threadIdx.x + o];
        __syncthreads();
    }
    float rstd = rsqrtf(red[0] / N + 1e-5f);
    __bf16* y = out + (size_t)row * N;
    for (int j = threadIdx.x; j < N; j += 256) {
        float xv = x[j];
        y[j] = f2bf(g[j] * (xv - mu) * rstd + be[j] + xv);
    }
}

__global__ __launch_bounds__(128)
void k_rownorm(const float* __restrict__ Cb, float* __restrict__ cn, int K) {
    __shared__ float red[128];
    int row = blockIdx.x;
    const float* c = Cb + (size_t)row * K;
    float s = 0.f;
    for (int j = threadIdx.x; j < K; j += 128) s += c[j] * c[j];
    red[threadIdx.x] = s; __syncthreads();
    for (int o = 64; o > 0; o >>= 1) {
        if (threadIdx.x < o) red[threadIdx.x] += red[threadIdx.x + o];
        __syncthreads();
    }
    if (threadIdx.x == 0) cn[row] = red[0];
}

// gather z_q = codebook[idx] (bf16 out) + code loss = (1+beta)*mean((zq-z)^2)
__global__ __launch_bounds__(256)
void k_vq_gather_loss(const unsigned long long* __restrict__ minbuf,
                      const float* __restrict__ z, const float* __restrict__ Cb,
                      __bf16* __restrict__ zq, float* __restrict__ loss,
                      int Kd, float scale) {
    __shared__ float red[256];
    int row = blockIdx.x;
    unsigned idx = (unsigned)(minbuf[row] & 0xFFFFFFFFu);
    const float* zr = z  + (size_t)row * Kd;
    const float* cr = Cb + (size_t)idx * Kd;
    float s = 0.f;
    for (int j = threadIdx.x; j < Kd; j += 256) {
        float c = cr[j];
        float d = c - zr[j];
        s += d * d;
        zq[(size_t)row * Kd + j] = f2bf(c);
    }
    red[threadIdx.x] = s; __syncthreads();
    for (int o = 128; o > 0; o >>= 1) {
        if (threadIdx.x < o) red[threadIdx.x] += red[threadIdx.x + o];
        __syncthreads();
    }
    if (threadIdx.x == 0) atomicAdd(loss, red[0] * scale);
}

__global__ void k_init(unsigned long long* minbuf, float* loss, int n) {
    int i = blockIdx.x * blockDim.x + threadIdx.x;
    if (i < n) minbuf[i] = ~0ull;
    if (i == 0) *loss = 0.f;
}

__global__ void k_convert(const float* __restrict__ s, __bf16* __restrict__ d, int n) {
    int i = blockIdx.x * blockDim.x + threadIdx.x;
    if (i < n) d[i] = f2bf(s[i]);
}

// W:[K][N] row-major -> Wt:[N][K] bf16
__global__ void k_convert_t(const float* __restrict__ W, __bf16* __restrict__ Wt,
                            int K, int N) {
    int i = blockIdx.x * blockDim.x + threadIdx.x;
    if (i < K * N) {
        int n = i / K, k = i % K;
        Wt[i] = f2bf(W[(size_t)k * N + n]);
    }
}

__global__ void k_concat(const float* __restrict__ dense, const int* __restrict__ tok,
                         const float* __restrict__ table, __bf16* __restrict__ X) {
    int b = blockIdx.x;
    int t = tok[b];
    __bf16* xr = X + (size_t)b * 1280;
    for (int j = threadIdx.x; j < 1024; j += 256)
        xr[j] = f2bf(dense[(size_t)b * 1024 + j]);
    for (int j = threadIdx.x; j < 256; j += 256)
        xr[1024 + j] = f2bf(table[(size_t)t * 256 + j]);
}

// ---------------------------------------------------------------------------
extern "C" void kernel_launch(void* const* d_in, const int* in_sizes, int n_in,
                              void* d_out, int out_size, void* d_ws, size_t ws_size,
                              hipStream_t stream) {
    const int B = 16384, CIN = 1280, EC_HID = 1024, EC_OUT = 512,
              CODE_N = 8192, DC_HID = 1024, TGT = 1024;
    (void)in_sizes; (void)n_in; (void)out_size; (void)ws_size;

    const float* dense = (const float*)d_in[0];
    const int*   tok   = (const int*)d_in[1];
    const float* table = (const float*)d_in[2];
    const float* eW1 = (const float*)d_in[3];  const float* eb1  = (const float*)d_in[4];
    const float* eg1 = (const float*)d_in[5];  const float* ebe1 = (const float*)d_in[6];
    const float* eW2 = (const float*)d_in[7];  const float* eb2  = (const float*)d_in[8];
    const float* cb  = (const float*)d_in[9];
    const float* dW1 = (const float*)d_in[10]; const float* db1  = (const float*)d_in[11];
    const float* dg1 = (const float*)d_in[12]; const float* dbe1 = (const float*)d_in[13];
    const float* dW2 = (const float*)d_in[14]; const float* db2  = (const float*)d_in[15];

    float* out  = (float*)d_out;
    float* loss = out + (size_t)B * TGT;

    char* ws = (char*)d_ws;
    size_t off = 0;
    auto alloc = [&](size_t bytes) -> char* {
        off = (off + 255) & ~(size_t)255;
        char* p = ws + off; off += bytes; return p;
    };
    __bf16* Wt1 = (__bf16*)alloc((size_t)CIN * EC_HID * 2);
    __bf16* Wt2 = (__bf16*)alloc((size_t)EC_HID * EC_OUT * 2);
    __bf16* Cbb = (__bf16*)alloc((size_t)CODE_N * EC_OUT * 2);
    __bf16* Wt3 = (__bf16*)alloc((size_t)EC_OUT * DC_HID * 2);
    __bf16* Wt4 = (__bf16*)alloc((size_t)DC_HID * TGT * 2);
    __bf16* Xb  = (__bf16*)alloc((size_t)B * CIN * 2);
    float*  Hf  = (float*)alloc((size_t)B * EC_HID * 4);
    __bf16* Hb  = (__bf16*)alloc((size_t)B * EC_HID * 2);
    float*  zf  = (float*)alloc((size_t)B * EC_OUT * 4);
    __bf16* zb  = (__bf16*)alloc((size_t)B * EC_OUT * 2);
    float*  cn  = (float*)alloc((size_t)CODE_N * 4);
    unsigned long long* mb = (unsigned long long*)alloc((size_t)B * 8);

    k_init<<<(B + 255) / 256, 256, 0, stream>>>(mb, loss, B);
    k_convert_t<<<(CIN * EC_HID + 255) / 256, 256, 0, stream>>>(eW1, Wt1, CIN, EC_HID);
    k_convert_t<<<(EC_HID * EC_OUT + 255) / 256, 256, 0, stream>>>(eW2, Wt2, EC_HID, EC_OUT);
    k_convert<<<(CODE_N * EC_OUT + 255) / 256, 256, 0, stream>>>(cb, Cbb, CODE_N * EC_OUT);
    k_convert_t<<<(EC_OUT * DC_HID + 255) / 256, 256, 0, stream>>>(dW1, Wt3, EC_OUT, DC_HID);
    k_convert_t<<<(DC_HID * TGT + 255) / 256, 256, 0, stream>>>(dW2, Wt4, DC_HID, TGT);
    k_concat<<<B, 256, 0, stream>>>(dense, tok, table, Xb);
    k_rownorm<<<CODE_N, 128, 0, stream>>>(cb, cn, EC_OUT);

    // encoder layer 1 + LN/residual
    k_gemm<false><<<dim3(EC_HID / BN, B / BM), TPB, 0, stream>>>(
        Xb, Wt1, eb1, Hf, nullptr, nullptr, B, EC_HID, CIN);
    k_ln_res<<<B, 256, 0, stream>>>(Hf, eg1, ebe1, Hb, EC_HID);
    // encoder layer 2 -> z
    k_gemm<false><<<dim3(EC_OUT / BN, B / BM), TPB, 0, stream>>>(
        Hb, Wt2, eb2, zf, nullptr, nullptr, B, EC_OUT, EC_HID);
    k_convert<<<(B * EC_OUT + 255) / 256, 256, 0, stream>>>(zf, zb, B * EC_OUT);
    // VQ: fused distance GEMM + argmin
    k_gemm<true><<<dim3(CODE_N / BN, B / BM), TPB, 0, stream>>>(
        zb, Cbb, nullptr, nullptr, mb, cn, B, CODE_N, EC_OUT);
    k_vq_gather_loss<<<B, 256, 0, stream>>>(
        mb, zf, cb, zb, loss, EC_OUT, (1.0f + 0.1f) / ((float)B * EC_OUT));
    // decoder layer 1 + LN/residual (straight-through: input is z_q)
    k_gemm<false><<<dim3(DC_HID / BN, B / BM), TPB, 0, stream>>>(
        zb, Wt3, db1, Hf, nullptr, nullptr, B, DC_HID, EC_OUT);
    k_ln_res<<<B, 256, 0, stream>>>(Hf, dg1, dbe1, Hb, DC_HID);
    // decoder layer 2 -> out
    k_gemm<false><<<dim3(TGT / BN, B / BM), TPB, 0, stream>>>(
        Hb, Wt4, db2, out, nullptr, nullptr, B, TGT, DC_HID);
}
